// CrossAttention_46231027974204
// MI455X (gfx1250) — matrix operand order repaired and checked by reading, not compiled, not run on hardware
//
#include <hip/hip_runtime.h>
#include <hip/hip_bf16.h>

typedef __bf16 bf16_t;
typedef __bf16 v16bf __attribute__((ext_vector_type(16)));
typedef float  v8f   __attribute__((ext_vector_type(8)));

#define WMMA_BF16(a, b, c) \
  __builtin_amdgcn_wmma_f32_16x16x32_bf16(false, (a), false, (b), (short)0, (c), false, false)

// Problem constants (B=1)
#define NTOK   4096
#define DMODEL 640
#define INNER  512
#define NHEADS 8
#define DHEAD  64
// 1/sqrt(64) * log2(e): softmax computed with exp2 (v_exp_f32 is base-2 native)
#define ATTN_SCALE_LOG2E (0.125f * 1.44269504088896f)

__device__ inline v8f v8f_zero() {
  v8f z = {0.f, 0.f, 0.f, 0.f, 0.f, 0.f, 0.f, 0.f};
  return z;
}

// A-matrix fragment (16xK tile, bf16). ISA layout: lanes 0-15 = rows 0-15 with
// K elems {0..7, 16..23}; lanes 16-31 same rows with K elems {8..15, 24..31}.
__device__ inline v16bf load_a_frag(const bf16_t* base, int stride, int lane) {
  const bf16_t* p = base + (lane & 15) * stride + ((lane < 16) ? 0 : 8);
  union { v16bf v; float4 q[2]; } u;
  u.q[0] = *(const float4*)(p);
  u.q[1] = *(const float4*)(p + 16);
  return u.v;
}

// B-matrix fragment (32x16 tile, bf16) from [n][k]-major storage.
// ISA layout: lanes 0-15 = col n, K {0..7, 16..23}; lanes 16-31 = col n, K {8..15, 24..31}.
__device__ inline v16bf load_b_frag(const bf16_t* base, int stride, int lane) {
  const bf16_t* p = base + (lane & 15) * stride + ((lane < 16) ? 0 : 8);
  union { v16bf v; float4 q[2]; } u;
  u.q[0] = *(const float4*)(p);
  u.q[1] = *(const float4*)(p + 16);
  return u.v;
}

// ---------------------------------------------------------------------------
// f32 -> bf16 conversion (vectorized x4)
// ---------------------------------------------------------------------------
__global__ void cvt_f32_to_bf16(const float* __restrict__ src,
                                bf16_t* __restrict__ dst, int n4) {
  int i = blockIdx.x * blockDim.x + threadIdx.x;
  if (i >= n4) return;
  float4 f = ((const float4*)src)[i];
  union { bf16_t e[4]; uint2 u; } o;
  o.e[0] = (bf16_t)f.x; o.e[1] = (bf16_t)f.y;
  o.e[2] = (bf16_t)f.z; o.e[3] = (bf16_t)f.w;
  ((uint2*)dst)[i] = o.u;
}

// ---------------------------------------------------------------------------
// Tiled bf16 WMMA GEMM: C[M,N] = alpha * A[M,K] @ B[K,N]  (+ bias, f32 out)
// Block tile 128x64, 8 waves, wave tile 32x32, K step 32.
// ---------------------------------------------------------------------------
__global__ __launch_bounds__(256) void gemm_bf16_wmma(
    const bf16_t* __restrict__ A, const bf16_t* __restrict__ B,
    bf16_t* __restrict__ Cb, float* __restrict__ Cf,
    const float* __restrict__ bias, int M, int N, int K, float alpha) {
  __shared__ bf16_t Ald[128][40];  // [m][k], padded
  __shared__ bf16_t Bld[64][40];   // [n][k], padded (transposed on fill)

  const int tid  = threadIdx.x;
  const int lane = tid & 31;
  const int wave = tid >> 5;
  const int m0 = blockIdx.y * 128;
  const int n0 = blockIdx.x * 64;
  const int mw = (wave >> 1) * 32;
  const int nw = (wave & 1) * 32;

  v8f acc[2][2];
  acc[0][0] = v8f_zero(); acc[0][1] = v8f_zero();
  acc[1][0] = v8f_zero(); acc[1][1] = v8f_zero();

  const int kt_count = K >> 5;
  for (int kt = 0; kt < kt_count; ++kt) {
    const int k0 = kt << 5;
    // A tile: 128x32 bf16, row-major, coalesced 16B per thread x2
#pragma unroll
    for (int i = 0; i < 2; ++i) {
      int idx = tid + i * 256;
      int row = idx >> 2;
      int c8  = (idx & 3) * 8;
      *(float4*)&Ald[row][c8] =
          *(const float4*)(A + (size_t)(m0 + row) * K + k0 + c8);
    }
    // B tile: 32x64 read row-major, stored transposed [n][k]
    {
      int kr = tid >> 3;
      int n8 = (tid & 7) * 8;
      union { float4 f; bf16_t e[8]; } u;
      u.f = *(const float4*)(B + (size_t)(k0 + kr) * N + n0 + n8);
#pragma unroll
      for (int i = 0; i < 8; ++i) Bld[n8 + i][kr] = u.e[i];
    }
    __syncthreads();

    v16bf a0 = load_a_frag(&Ald[mw][0], 40, lane);
    v16bf a1 = load_a_frag(&Ald[mw + 16][0], 40, lane);
    v16bf b0 = load_b_frag(&Bld[nw][0], 40, lane);
    v16bf b1 = load_b_frag(&Bld[nw + 16][0], 40, lane);
    acc[0][0] = WMMA_BF16(a0, b0, acc[0][0]);
    acc[0][1] = WMMA_BF16(a0, b1, acc[0][1]);
    acc[1][0] = WMMA_BF16(a1, b0, acc[1][0]);
    acc[1][1] = WMMA_BF16(a1, b1, acc[1][1]);
    __syncthreads();
  }

  // Epilogue. C layout: VGPR r, lanes 0-15 -> row r, lanes 16-31 -> row r+8.
  const int rowoff = (lane < 16) ? 0 : 8;
  const int col    = lane & 15;
#pragma unroll
  for (int i = 0; i < 2; ++i)
#pragma unroll
    for (int j = 0; j < 2; ++j)
#pragma unroll
      for (int r = 0; r < 8; ++r) {
        int gr = m0 + mw + i * 16 + rowoff + r;
        int gc = n0 + nw + j * 16 + col;
        float v = alpha * acc[i][j][r];
        if (Cb) Cb[(size_t)gr * N + gc] = (bf16_t)v;
        else    Cf[(size_t)gr * N + gc] = v + bias[gc];
      }
}

// ---------------------------------------------------------------------------
// Flash attention: one head column (d=64), 128 query rows / block (16 / wave),
// 64-key tiles with register-staged prefetch of the next tile.
// Q pre-scaled by (1/sqrt(d))*log2(e); softmax uses exp2. Buffers bf16.
// ---------------------------------------------------------------------------
__global__ __launch_bounds__(256) void flash_attn_wmma(
    const bf16_t* __restrict__ Q, const bf16_t* __restrict__ Km,
    const bf16_t* __restrict__ Vm, bf16_t* __restrict__ O) {
  __shared__ bf16_t Kld[64][72];      // [key][d]  (B-frag source for Q@K^T)
  __shared__ bf16_t Vld[64][72];      // [d][key]  (B-frag source for P@V)
  __shared__ bf16_t Pld[8][16][72];   // per-wave P re-layout buffer

  const int tid  = threadIdx.x;
  const int lane = tid & 31;
  const int wave = tid >> 5;
  const int head = blockIdx.y;
  const int q0   = blockIdx.x * 128 + wave * 16;
  const int hoff = head * DHEAD;
  const int rowoff = (lane < 16) ? 0 : 8;
  const int col    = lane & 15;

  // Q tile (16x64) stays in registers: two A-fragments (d 0-31, d 32-63)
  v16bf aq0 = load_a_frag(Q + (size_t)q0 * INNER + hoff + 0,  INNER, lane);
  v16bf aq1 = load_a_frag(Q + (size_t)q0 * INNER + hoff + 32, INNER, lane);

  v8f oacc[4];
  oacc[0] = v8f_zero(); oacc[1] = v8f_zero();
  oacc[2] = v8f_zero(); oacc[3] = v8f_zero();
  float Mrow[8], Lrow[8];
#pragma unroll
  for (int r = 0; r < 8; ++r) { Mrow[r] = -3.0e38f; Lrow[r] = 0.0f; }

  // tile-fill mapping: thread covers key row kr, dim chunk [c0, c0+16)
  const int kr = tid >> 2;
  const int c0 = (tid & 3) * 16;

  float4 pk0, pk1;                 // staged K regs (16 bf16)
  union { float4 f[2]; bf16_t e[16]; } pv;  // staged V regs

  {
    const bf16_t* kp = Km + (size_t)kr * INNER + hoff + c0;
    const bf16_t* vp = Vm + (size_t)kr * INNER + hoff + c0;
    pk0 = *(const float4*)kp; pk1 = *(const float4*)(kp + 8);
    pv.f[0] = *(const float4*)vp; pv.f[1] = *(const float4*)(vp + 8);
  }

  for (int kt = 0; kt < NTOK / 64; ++kt) {
    __syncthreads();  // previous tile's LDS fully consumed
    // ---- commit staged tile to LDS (K row-major, V transposed) ----
    *(float4*)&Kld[kr][c0]     = pk0;
    *(float4*)&Kld[kr][c0 + 8] = pk1;
#pragma unroll
    for (int i = 0; i < 16; ++i) Vld[c0 + i][kr] = pv.e[i];
    __syncthreads();

    // ---- prefetch next tile into registers (overlaps compute below) ----
    {
      const int ktn = (kt + 1) & (NTOK / 64 - 1);  // last iter: harmless reload
      const size_t src = (size_t)(ktn * 64 + kr) * INNER + hoff + c0;
      pk0 = *(const float4*)(Km + src);
      pk1 = *(const float4*)(Km + src + 8);
      pv.f[0] = *(const float4*)(Vm + src);
      pv.f[1] = *(const float4*)(Vm + src + 8);
    }

    // ---- S = Q @ K^T  (16x64, four 16x16 accumulators) ----
    v8f s[4];
#pragma unroll
    for (int j = 0; j < 4; ++j) {
      s[j] = v8f_zero();
      v16bf bk0 = load_b_frag(&Kld[j * 16][0],  72, lane);
      v16bf bk1 = load_b_frag(&Kld[j * 16][32], 72, lane);
      s[j] = WMMA_BF16(aq0, bk0, s[j]);
      s[j] = WMMA_BF16(aq1, bk1, s[j]);
    }

    // ---- online softmax (base-2), row reductions across 16-lane halves ----
    float alr[8];
#pragma unroll
    for (int r = 0; r < 8; ++r) {
      float t = fmaxf(fmaxf(s[0][r], s[1][r]), fmaxf(s[2][r], s[3][r]));
      t = fmaxf(t, __shfl_xor(t, 1));
      t = fmaxf(t, __shfl_xor(t, 2));
      t = fmaxf(t, __shfl_xor(t, 4));
      t = fmaxf(t, __shfl_xor(t, 8));
      float mnew = fmaxf(Mrow[r], t);
      float al   = exp2f(Mrow[r] - mnew);
      Mrow[r] = mnew;
      float p0 = exp2f(s[0][r] - mnew);
      float p1 = exp2f(s[1][r] - mnew);
      float p2 = exp2f(s[2][r] - mnew);
      float p3 = exp2f(s[3][r] - mnew);
      s[0][r] = p0; s[1][r] = p1; s[2][r] = p2; s[3][r] = p3;
      float sum = (p0 + p1) + (p2 + p3);
      sum += __shfl_xor(sum, 1);
      sum += __shfl_xor(sum, 2);
      sum += __shfl_xor(sum, 4);
      sum += __shfl_xor(sum, 8);
      Lrow[r] = Lrow[r] * al + sum;
      alr[r] = al;
    }
#pragma unroll
    for (int b = 0; b < 4; ++b)
#pragma unroll
      for (int r = 0; r < 8; ++r) oacc[b][r] *= alr[r];

    // ---- P (C layout) -> per-wave LDS -> A-fragments; O += P @ V ----
    bf16_t* pw = &Pld[wave][0][0];
#pragma unroll
    for (int r = 0; r < 8; ++r) {
#pragma unroll
      for (int j = 0; j < 4; ++j)
        pw[(rowoff + r) * 72 + j * 16 + col] = (bf16_t)s[j][r];
    }
    v16bf pa0 = load_a_frag(pw,      72, lane);  // keys 0-31
    v16bf pa1 = load_a_frag(pw + 32, 72, lane);  // keys 32-63
#pragma unroll
    for (int b = 0; b < 4; ++b) {
      v16bf vb0 = load_b_frag(&Vld[b * 16][0],  72, lane);
      v16bf vb1 = load_b_frag(&Vld[b * 16][32], 72, lane);
      oacc[b] = WMMA_BF16(pa0, vb0, oacc[b]);
      oacc[b] = WMMA_BF16(pa1, vb1, oacc[b]);
    }
  }

  // ---- normalize and store O (bf16) ----
#pragma unroll
  for (int b = 0; b < 4; ++b)
#pragma unroll
    for (int r = 0; r < 8; ++r) {
      float v = oacc[b][r] / Lrow[r];
      O[(size_t)(q0 + rowoff + r) * INNER + hoff + b * 16 + col] = (bf16_t)v;
    }
}

// ---------------------------------------------------------------------------
// Host launcher
// ---------------------------------------------------------------------------
extern "C" void kernel_launch(void* const* d_in, const int* in_sizes, int n_in,
                              void* d_out, int out_size, void* d_ws,
                              size_t ws_size, hipStream_t stream) {
  (void)in_sizes; (void)n_in; (void)out_size; (void)ws_size;
  const float* x  = (const float*)d_in[0];
  const float* Wq = (const float*)d_in[1];
  const float* Wk = (const float*)d_in[2];
  const float* Wv = (const float*)d_in[3];
  const float* Wo = (const float*)d_in[4];
  const float* bo = (const float*)d_in[5];

  char* ws = (char*)d_ws;
  auto carve = [&](size_t bytes) {
    void* p = (void*)ws;
    ws += (bytes + 255) & ~(size_t)255;
    return p;
  };
  bf16_t* xb  = (bf16_t*)carve((size_t)NTOK * DMODEL * 2);
  bf16_t* wqb = (bf16_t*)carve((size_t)DMODEL * INNER * 2);
  bf16_t* wkb = (bf16_t*)carve((size_t)DMODEL * INNER * 2);
  bf16_t* wvb = (bf16_t*)carve((size_t)DMODEL * INNER * 2);
  bf16_t* wob = (bf16_t*)carve((size_t)INNER * DMODEL * 2);
  bf16_t* Qb  = (bf16_t*)carve((size_t)NTOK * INNER * 2);
  bf16_t* Kb  = (bf16_t*)carve((size_t)NTOK * INNER * 2);
  bf16_t* Vb  = (bf16_t*)carve((size_t)NTOK * INNER * 2);
  bf16_t* Ob  = (bf16_t*)carve((size_t)NTOK * INNER * 2);

  auto cvt = [&](const float* s, bf16_t* d, int n) {
    int n4 = n / 4;
    cvt_f32_to_bf16<<<(n4 + 255) / 256, 256, 0, stream>>>(s, d, n4);
  };
  cvt(x,  xb,  NTOK * DMODEL);
  cvt(Wq, wqb, DMODEL * INNER);
  cvt(Wk, wkb, DMODEL * INNER);
  cvt(Wv, wvb, DMODEL * INNER);
  cvt(Wo, wob, INNER * DMODEL);

  // QKV projections (Q pre-scaled by (1/sqrt(d))*log2(e) for exp2 softmax)
  dim3 gP(INNER / 64, NTOK / 128);
  gemm_bf16_wmma<<<gP, 256, 0, stream>>>(xb, wqb, Qb, nullptr, nullptr,
                                         NTOK, INNER, DMODEL, ATTN_SCALE_LOG2E);
  gemm_bf16_wmma<<<gP, 256, 0, stream>>>(xb, wkb, Kb, nullptr, nullptr,
                                         NTOK, INNER, DMODEL, 1.0f);
  gemm_bf16_wmma<<<gP, 256, 0, stream>>>(xb, wvb, Vb, nullptr, nullptr,
                                         NTOK, INNER, DMODEL, 1.0f);

  // Fused attention
  flash_attn_wmma<<<dim3(NTOK / 128, NHEADS), 256, 0, stream>>>(Qb, Kb, Vb, Ob);

  // Output projection + bias (f32 out)
  dim3 gO(DMODEL / 64, NTOK / 128);
  gemm_bf16_wmma<<<gO, 256, 0, stream>>>(Ob, wob, nullptr, (float*)d_out, bo,
                                         NTOK, DMODEL, INNER, 1.0f);
}
